// VariableSizeExpertLayer_12893491823133
// MI455X (gfx1250) — compile-verified, matrix operand order checked by reading, not run on hardware
//
#include <hip/hip_runtime.h>
#include <hip/hip_bf16.h>
#include <cstdint>
#include <cstddef>

// ---------------------------------------------------------------------------
// Top-2 MoE FFN for MI455X (gfx1250, wave32, WMMA).
// Routed (not dense) computation: 0.26 TFLOP via v_wmma_f32_16x16x32_bf16.
// Workspace layout (bytes):
//   [0)      int counts[8]
//   [64)     int hoff[8]          (bf16-element offsets into h buffer)
//   [1024)   int topi[16384*2]
//   [132096) float topw[16384*2]
//   [263168) int lists[8*16384]
//   [787456) float gw[8*16384]
//   [1311744) bf16 h buffer       (compact: sum_e count[e]*ff[e] elements)
// ---------------------------------------------------------------------------

typedef __bf16 bf16_t;
typedef __attribute__((ext_vector_type(16))) __bf16 v16bf;
typedef __attribute__((ext_vector_type(8)))  float  v8f;

#define NEXP 8
#define HID  1024
#define NTOK 16384
#define CAP  16384

#define WS_COUNTS 0
#define WS_HOFF   64
#define WS_TOPI   1024
#define WS_TOPW   132096
#define WS_LISTS  263168
#define WS_GW     787456
#define WS_H      1311744

struct Ptr8 { const float* p[8]; };

union Pack16 { bf16_t h[16]; uint4 u[2]; };

__device__ __forceinline__ float gelu_exact(float v) {
  return 0.5f * v * (1.0f + erff(v * 0.70710678118654752f));
}

__device__ __forceinline__ void pack16(Pack16& p, float4 a, float4 b, float4 c, float4 d) {
  p.h[0]=(bf16_t)a.x;  p.h[1]=(bf16_t)a.y;  p.h[2]=(bf16_t)a.z;  p.h[3]=(bf16_t)a.w;
  p.h[4]=(bf16_t)b.x;  p.h[5]=(bf16_t)b.y;  p.h[6]=(bf16_t)b.z;  p.h[7]=(bf16_t)b.w;
  p.h[8]=(bf16_t)c.x;  p.h[9]=(bf16_t)c.y;  p.h[10]=(bf16_t)c.z; p.h[11]=(bf16_t)c.w;
  p.h[12]=(bf16_t)d.x; p.h[13]=(bf16_t)d.y; p.h[14]=(bf16_t)d.z; p.h[15]=(bf16_t)d.w;
}

// ---------------- 1) gating: logits, top-2, softmax ----------------
__global__ __launch_bounds__(256) void moe_gate(
    const float* __restrict__ x, const float* __restrict__ gate_w,
    const float* __restrict__ gate_b, int* __restrict__ topi,
    float* __restrict__ topw)
{
  __shared__ float gl[HID * NEXP];           // 32 KB: gate_w [h][e]
  int tid = threadIdx.x;
  for (int i = tid; i < HID * NEXP; i += 256) gl[i] = gate_w[i];
  __syncthreads();

  int t = blockIdx.x * 256 + tid;
  float acc[8];
#pragma unroll
  for (int e = 0; e < 8; ++e) acc[e] = gate_b[e];

  const float4* xr = (const float4*)(x + (size_t)t * HID);
  for (int h = 0; h < HID; h += 4) {
    float4 v = xr[h >> 2];
    const float* g = &gl[h * 8];
#pragma unroll
    for (int e = 0; e < 8; ++e)
      acc[e] += v.x * g[e] + v.y * g[8 + e] + v.z * g[16 + e] + v.w * g[24 + e];
  }
  // top-2, ties -> lower index (matches jax.lax.top_k)
  int i0 = 0; float v0 = acc[0];
#pragma unroll
  for (int e = 1; e < 8; ++e) if (acc[e] > v0) { v0 = acc[e]; i0 = e; }
  int i1 = (i0 == 0) ? 1 : 0; float v1 = acc[i1];
#pragma unroll
  for (int e = 0; e < 8; ++e)
    if (e != i0 && acc[e] > v1) { v1 = acc[e]; i1 = e; }

  float e1  = __expf(v1 - v0);               // v0 >= v1 -> stable softmax
  float inv = 1.0f / (1.0f + e1);
  topi[t * 2] = i0;  topi[t * 2 + 1] = i1;
  topw[t * 2] = inv; topw[t * 2 + 1] = e1 * inv;
}

// ---------------- 2) stable per-expert compaction (deterministic) ----------
__global__ __launch_bounds__(256) void moe_build(
    const int* __restrict__ topi, const float* __restrict__ topw,
    int* __restrict__ counts, int* __restrict__ hoffp,
    int* __restrict__ lists, float* __restrict__ gw)
{
  __shared__ int scount[8];
  int wave = threadIdx.x >> 5, lane = threadIdx.x & 31;
  int e = wave;                              // 8 waves, one expert each
  int base = 0;
  for (int t0 = 0; t0 < NTOK; t0 += 32) {
    int t = t0 + lane;
    int a = topi[t * 2], b = topi[t * 2 + 1];
    bool hit = (a == e) || (b == e);
    float w = (a == e) ? topw[t * 2] : topw[t * 2 + 1];
    unsigned mask = (unsigned)__ballot(hit); // wave32: low 32 bits
    if (hit) {
      int pos = base + __popc(mask & ((1u << lane) - 1u));
      lists[e * CAP + pos] = t;
      gw[e * CAP + pos] = w;
    }
    base += __popc(mask);
  }
  if (lane == 0) scount[e] = base;
  __syncthreads();
  if (threadIdx.x == 0) {
    const int ffd[8] = {1024, 1536, 2048, 3072, 3072, 2048, 1536, 1024};
    int off = 0;
#pragma unroll
    for (int i = 0; i < 8; ++i) {
      counts[i] = scount[i];
      hoffp[i] = off;
      off += scount[i] * ffd[i];
    }
  }
}

// ---------------- 3) GEMM1 + GELU: h = gelu(Xg @ W1 + b1), bf16 out --------
// Block tile 128x128, K-step 32. 8 waves: 2(m) x 4(n); wave tile 64x32 =
// 4x2 WMMA 16x16 accumulators. A/B staged in LDS as bf16, WMMA lane layout:
//   A 16x32: lane(0-15)=row M, k 0..7 & 16..23; lane(16-31): k 8..15 & 24..31
//   B 32x16: mirrored with N on lanes. Fragments = two ds_load_b128 each.
__global__ __launch_bounds__(256) void moe_ffn1(
    const float* __restrict__ x, Ptr8 w1, Ptr8 b1,
    const int* __restrict__ counts, const int* __restrict__ hoffp,
    const int* __restrict__ lists, bf16_t* __restrict__ hbuf)
{
  const int ffd[8] = {1024, 1536, 2048, 3072, 3072, 2048, 1536, 1024};
  int rem = blockIdx.x, e = 0;
  while (rem >= (ffd[e] >> 7)) { rem -= ffd[e] >> 7; ++e; }
  int cnt = counts[e];
  int row0 = blockIdx.y << 7;
  if (row0 >= cnt) return;                   // uniform early exit
  int ff = ffd[e];
  int colbase = rem << 7;
  const float* __restrict__ W    = w1.p[e];
  const float* __restrict__ bias = b1.p[e];

  __shared__ bf16_t As[128 * 48];            // stride 48 bf16 (96B, 16B-mult)
  __shared__ bf16_t Bs[128 * 48];
  __shared__ int tok_s[128];

  int tid = threadIdx.x;
  if (tid < 128) {
    int r = (row0 + tid < cnt) ? tid : 0;    // clamp to a valid row
    tok_s[tid] = lists[e * CAP + row0 + r];
  }
  __syncthreads();

  int sm = tid & 127;                        // staging row (A) / col (B)
  int sk = (tid >> 7) << 4;                  // k-chunk: 0 or 16
  const float* xrow  = x + (size_t)tok_s[sm] * HID + sk;
  const float* wbase = W + colbase + sm + (size_t)sk * ff;

  int lane = tid & 31, wave = tid >> 5;
  int wm = wave >> 2, wn = wave & 3;
  int lr = lane & 15, lh = lane >> 4;

  v8f acc[4][2] = {};

  for (int k0 = 0; k0 < HID; k0 += 32) {
    // stage A (gathered x rows -> bf16)
    const float4* srcA = (const float4*)(xrow + k0);
    Pack16 pa; pack16(pa, srcA[0], srcA[1], srcA[2], srcA[3]);
    *(uint4*)&As[sm * 48 + sk]     = pa.u[0];
    *(uint4*)&As[sm * 48 + sk + 8] = pa.u[1];
    // stage B transposed: LDS[n][k] <- W1[k][n]
    Pack16 pb;
    {
      const float* wp = wbase + (size_t)k0 * ff;
#pragma unroll
      for (int i = 0; i < 16; ++i) pb.h[i] = (bf16_t)wp[(size_t)i * ff];
    }
    *(uint4*)&Bs[sm * 48 + sk]     = pb.u[0];
    *(uint4*)&Bs[sm * 48 + sk + 8] = pb.u[1];
    __syncthreads();

    v16bf afr[4], bfr[2];
#pragma unroll
    for (int mt = 0; mt < 4; ++mt) {
      int m = wm * 64 + mt * 16 + lr;
      union { uint4 u[2]; v16bf v; } fa;
      fa.u[0] = *(const uint4*)&As[m * 48 + lh * 8];
      fa.u[1] = *(const uint4*)&As[m * 48 + lh * 8 + 16];
      afr[mt] = fa.v;
    }
#pragma unroll
    for (int nt = 0; nt < 2; ++nt) {
      int n = wn * 32 + nt * 16 + lr;
      union { uint4 u[2]; v16bf v; } fb;
      fb.u[0] = *(const uint4*)&Bs[n * 48 + lh * 8];
      fb.u[1] = *(const uint4*)&Bs[n * 48 + lh * 8 + 16];
      bfr[nt] = fb.v;
    }
#pragma unroll
    for (int mt = 0; mt < 4; ++mt)
#pragma unroll
      for (int nt = 0; nt < 2; ++nt)
        acc[mt][nt] = __builtin_amdgcn_wmma_f32_16x16x32_bf16(
            false, afr[mt], false, bfr[nt], (short)0, acc[mt][nt], false, false);
    __syncthreads();
  }

  // epilogue: +b1, GELU, bf16 store. D layout: VGPR r -> M = r + 8*lh, N = lr
  int hbase = hoffp[e];
#pragma unroll
  for (int mt = 0; mt < 4; ++mt) {
#pragma unroll
    for (int nt = 0; nt < 2; ++nt) {
      int col = colbase + wn * 32 + nt * 16 + lr;
      float bv = bias[col];
#pragma unroll
      for (int r = 0; r < 8; ++r) {
        int mrow = wm * 64 + mt * 16 + lh * 8 + r;
        if (row0 + mrow < cnt) {
          float v = acc[mt][nt][r] + bv;
          hbuf[(size_t)hbase + (size_t)(row0 + mrow) * ff + col] =
              (bf16_t)gelu_exact(v);
        }
      }
    }
  }
}

// ---------------- 4) GEMM2 + scatter: out[tok] += gw * (h @ W2 + b2) -------
// Exactly 2 commutative float atomic adds per output element -> deterministic.
__global__ __launch_bounds__(256) void moe_ffn2(
    const bf16_t* __restrict__ hbuf, Ptr8 w2, Ptr8 b2,
    const int* __restrict__ counts, const int* __restrict__ hoffp,
    const int* __restrict__ lists, const float* __restrict__ gw,
    float* __restrict__ out)
{
  const int ffd[8] = {1024, 1536, 2048, 3072, 3072, 2048, 1536, 1024};
  int e = blockIdx.x >> 3;
  int colbase = (blockIdx.x & 7) << 7;
  int cnt = counts[e];
  int row0 = blockIdx.y << 7;
  if (row0 >= cnt) return;
  int ff = ffd[e];
  const float* __restrict__ W    = w2.p[e];
  const float* __restrict__ bias = b2.p[e];

  __shared__ bf16_t As[128 * 48];
  __shared__ bf16_t Bs[128 * 48];
  __shared__ int   tok_s[128];
  __shared__ float gw_s[128];

  int tid = threadIdx.x;
  if (tid < 128) {
    int r = (row0 + tid < cnt) ? tid : 0;
    tok_s[tid] = lists[e * CAP + row0 + r];
    gw_s[tid]  = gw[e * CAP + row0 + r];
  }
  __syncthreads();

  int sm = tid & 127;
  int sk = (tid >> 7) << 4;
  int srow = (row0 + sm < cnt) ? sm : 0;
  const bf16_t* hrow  = hbuf + (size_t)hoffp[e] + (size_t)(row0 + srow) * ff + sk;
  const float*  wbase = W + colbase + sm + (size_t)sk * HID;

  int lane = tid & 31, wave = tid >> 5;
  int wm = wave >> 2, wn = wave & 3;
  int lr = lane & 15, lh = lane >> 4;

  v8f acc[4][2] = {};

  for (int k0 = 0; k0 < ff; k0 += 32) {
    // stage A: h already bf16 -> straight b128 copies
    uint4 ha0 = *(const uint4*)(hrow + k0);
    uint4 ha1 = *(const uint4*)(hrow + k0 + 8);
    *(uint4*)&As[sm * 48 + sk]     = ha0;
    *(uint4*)&As[sm * 48 + sk + 8] = ha1;
    // stage B transposed: LDS[n][k] <- W2[k][n]
    Pack16 pb;
    {
      const float* wp = wbase + (size_t)k0 * HID;
#pragma unroll
      for (int i = 0; i < 16; ++i) pb.h[i] = (bf16_t)wp[(size_t)i * HID];
    }
    *(uint4*)&Bs[sm * 48 + sk]     = pb.u[0];
    *(uint4*)&Bs[sm * 48 + sk + 8] = pb.u[1];
    __syncthreads();

    v16bf afr[4], bfr[2];
#pragma unroll
    for (int mt = 0; mt < 4; ++mt) {
      int m = wm * 64 + mt * 16 + lr;
      union { uint4 u[2]; v16bf v; } fa;
      fa.u[0] = *(const uint4*)&As[m * 48 + lh * 8];
      fa.u[1] = *(const uint4*)&As[m * 48 + lh * 8 + 16];
      afr[mt] = fa.v;
    }
#pragma unroll
    for (int nt = 0; nt < 2; ++nt) {
      int n = wn * 32 + nt * 16 + lr;
      union { uint4 u[2]; v16bf v; } fb;
      fb.u[0] = *(const uint4*)&Bs[n * 48 + lh * 8];
      fb.u[1] = *(const uint4*)&Bs[n * 48 + lh * 8 + 16];
      bfr[nt] = fb.v;
    }
#pragma unroll
    for (int mt = 0; mt < 4; ++mt)
#pragma unroll
      for (int nt = 0; nt < 2; ++nt)
        acc[mt][nt] = __builtin_amdgcn_wmma_f32_16x16x32_bf16(
            false, afr[mt], false, bfr[nt], (short)0, acc[mt][nt], false, false);
    __syncthreads();
  }

#pragma unroll
  for (int mt = 0; mt < 4; ++mt) {
#pragma unroll
    for (int nt = 0; nt < 2; ++nt) {
      int col = colbase + wn * 32 + nt * 16 + lr;
      float bv = bias[col];
#pragma unroll
      for (int r = 0; r < 8; ++r) {
        int mrow = wm * 64 + mt * 16 + lh * 8 + r;
        if (row0 + mrow < cnt) {
          float v = (acc[mt][nt][r] + bv) * gw_s[mrow];
          unsafeAtomicAdd(&out[(size_t)tok_s[mrow] * HID + col], v);
        }
      }
    }
  }
}

// ---------------------------------------------------------------------------
extern "C" void kernel_launch(void* const* d_in, const int* in_sizes, int n_in,
                              void* d_out, int out_size, void* d_ws, size_t ws_size,
                              hipStream_t stream) {
  (void)in_sizes; (void)n_in; (void)ws_size;
  // setup_inputs order: x, gate_w, gate_b, w1s[0..7], b1s[0..7], w2s[0..7], b2s[0..7]
  const float* x      = (const float*)d_in[0];
  const float* gate_w = (const float*)d_in[1];
  const float* gate_b = (const float*)d_in[2];
  Ptr8 w1, b1, w2, b2;
  for (int i = 0; i < 8; ++i) {
    w1.p[i] = (const float*)d_in[3 + i];
    b1.p[i] = (const float*)d_in[11 + i];
    w2.p[i] = (const float*)d_in[19 + i];
    b2.p[i] = (const float*)d_in[27 + i];
  }
  char* ws = (char*)d_ws;
  int*    counts = (int*)(ws + WS_COUNTS);
  int*    hoffp  = (int*)(ws + WS_HOFF);
  int*    topi   = (int*)(ws + WS_TOPI);
  float*  topw   = (float*)(ws + WS_TOPW);
  int*    lists  = (int*)(ws + WS_LISTS);
  float*  gwv    = (float*)(ws + WS_GW);
  bf16_t* hbuf   = (bf16_t*)(ws + WS_H);
  float*  out    = (float*)d_out;

  hipMemsetAsync(out, 0, (size_t)out_size * sizeof(float), stream);
  moe_gate <<<dim3(NTOK / 256), 256, 0, stream>>>(x, gate_w, gate_b, topi, topw);
  moe_build<<<dim3(1), 256, 0, stream>>>(topi, topw, counts, hoffp, lists, gwv);
  // 120 col-tiles total over 8 experts (ff/128), 128 row-tiles max (early exit)
  moe_ffn1 <<<dim3(120, 128), 256, 0, stream>>>(x, w1, b1, counts, hoffp, lists, hbuf);
  // 8 experts x 8 col-tiles of HID=1024
  moe_ffn2 <<<dim3(64, 128), 256, 0, stream>>>(hbuf, w2, b2, counts, hoffp, lists, gwv, out);
}
